// MHSA_73186242724255
// MI455X (gfx1250) — matrix operand order repaired and checked by reading, not compiled, not run on hardware
//
#include <hip/hip_runtime.h>
#include <hip/hip_bf16.h>

#define B_  2
#define S_  2048
#define D_  1024
#define H_  16
#define HD_ 64
#define BS_ (B_*S_)   // 4096

typedef __attribute__((ext_vector_type(16))) __bf16 v16bf;
typedef __attribute__((ext_vector_type(8)))  __bf16 v8bf;   // 16 bytes
typedef __attribute__((ext_vector_type(8)))  float  v8f;

static __device__ __forceinline__ __bf16 to_bf16(float f) {
  unsigned int u = __builtin_bit_cast(unsigned int, f);
  u += 0x7FFFu + ((u >> 16) & 1u);                 // round-to-nearest-even
  unsigned short hs = (unsigned short)(u >> 16);
  return __builtin_bit_cast(__bf16, hs);
}

static __device__ __forceinline__ v8f zero8() {
  v8f z;
#pragma unroll
  for (int i = 0; i < 8; ++i) z[i] = 0.f;
  return z;
}

// Async 16B global -> LDS transfer (per-lane addresses, tracked by ASYNCcnt).
#define ASYNC_B128(ldsOff, gptr)                                            \
  asm volatile("global_load_async_to_lds_b128 %0, %1, off"                  \
               :: "v"(ldsOff), "v"(gptr) : "memory")
#define WAIT_ASYNC(n)  asm volatile("s_wait_asynccnt " #n ::: "memory")
#define DRAIN_DS()     asm volatile("s_wait_dscnt 0x0" ::: "memory")

// A fragment (16x32 bf16), CDNA5 ISA 7.12.2 layout: lane (m = L&15, h = L>>4)
// holds k = [8h .. 8h+7] then [16+8h .. 16+8h+7]  -> two contiguous 16B chunks.
static __device__ __forceinline__ v16bf load_a_frag(const __bf16* __restrict__ tile, int ld) {
  const int lane = threadIdx.x & 31;
  const int m = lane & 15;
  const int h = lane >> 4;
  const __bf16* p = tile + m*ld + 8*h;
  union { v16bf v; v8bf c[2]; } u;
  u.c[0] = *(const v8bf*)(p);
  u.c[1] = *(const v8bf*)(p + 16);
  return u.v;
}

// B fragment (32x16 bf16): lane (n = L&15, h = L>>4) holds k = [8h .. 8h+7] then [16+8h .. 16+8h+7] of column n.
static __device__ __forceinline__ v16bf load_b_frag(const __bf16* __restrict__ tile, int ld, int n0) {
  const int lane = threadIdx.x & 31;
  const int n = lane & 15;
  const int h = lane >> 4;
  const __bf16* p = tile + (8*h)*ld + n0 + n;
  v16bf v;
#pragma unroll
  for (int s = 0; s < 8; ++s) {
    v[s] = p[s*ld];
    v[8 + s] = p[(16 + s)*ld];
  }
  return v;
}

__global__ __launch_bounds__(256) void cvt_f32_bf16(const float* __restrict__ in,
                                                    __bf16* __restrict__ out, int n) {
  int i = blockIdx.x * 256 + threadIdx.x;
  if (i < n) out[i] = to_bf16(in[i]);
}

// W[out,in] f32  ->  Wt[in][out] bf16   (32x32 tiles via LDS)
__global__ __launch_bounds__(256) void cvt_transpose_bf16(const float* __restrict__ in,
                                                          __bf16* __restrict__ outT) {
  __shared__ float tile[32][33];
  const int bx = blockIdx.x * 32;           // in-dim (d) block
  const int by = blockIdx.y * 32;           // out-dim (c) block
  const int tx = threadIdx.x & 31;
  const int ty = threadIdx.x >> 5;          // 0..7
#pragma unroll
  for (int yy = ty; yy < 32; yy += 8)
    tile[yy][tx] = in[(size_t)(by + yy)*D_ + bx + tx];
  __syncthreads();
#pragma unroll
  for (int yy = ty; yy < 32; yy += 8)
    outT[(size_t)(bx + yy)*D_ + by + tx] = to_bf16(tile[tx][yy]);
}

// C[4096,1024] = A[4096,1024] * Wt([in][out]) + bias   (bf16 operands, f32 accum)
// Double-buffered async-DMA k-pipeline: tile t+1 streams into LDS while tile t
// runs on the WMMA pipes.
// mode 0: f32 row-major -> outF
// mode 1: bf16 -> [B,H,S,HD]   (Q / V)
// mode 2: bf16 -> [B,H,HD,S]   (K transposed)
#define LDA_ 40     // 32 + 8 pad (80B rows: 16B aligned, conflict-reduced)
#define LDW_ 136    // 128 + 8 pad (272B rows)
__global__ __launch_bounds__(256) void gemm_qkv(const __bf16* __restrict__ A,
                                                const __bf16* __restrict__ Wt,
                                                const float* __restrict__ bias,
                                                float* __restrict__ outF,
                                                __bf16* __restrict__ outB, int mode) {
  __shared__ __attribute__((aligned(16))) __bf16 ldsA[2][128*LDA_];
  __shared__ __attribute__((aligned(16))) __bf16 ldsW[2][32*LDW_];
  const int tid  = threadIdx.x;
  const int lane = tid & 31;
  const int w    = tid >> 5;          // 8 waves
  const int wm   = w & 3, wn = w >> 2;
  const int r0   = blockIdx.x * 128;
  const int c0   = blockIdx.y * 128;

  unsigned offA[2], offW[2];
#pragma unroll
  for (int s = 0; s < 2; ++s) {
    offA[s] = (unsigned)(size_t)(void*)&ldsA[s][0];
    offW[s] = (unsigned)(size_t)(void*)&ldsW[s][0];
  }

  // 4 async transfers per thread per tile: 2 for A (128x32), 2 for W (32x128).
  auto issue_tile = [&](int kb, int s) {
#pragma unroll
    for (int it = 0; it < 2; ++it) {
      const int c = tid + it*256;                 // 0..511
      const int ra = c >> 2, ja = c & 3;
      const unsigned la = offA[s] + (unsigned)((ra*LDA_ + ja*8)*2);
      const __bf16* ga = A + (size_t)(r0 + ra)*D_ + kb + ja*8;
      ASYNC_B128(la, ga);
      const int kw = c >> 4, jw = c & 15;
      const unsigned lw = offW[s] + (unsigned)((kw*LDW_ + jw*8)*2);
      const __bf16* gw = Wt + (size_t)(kb + kw)*D_ + c0 + jw*8;
      ASYNC_B128(lw, gw);
    }
  };

  v8f acc[2][4];
#pragma unroll
  for (int mi = 0; mi < 2; ++mi)
#pragma unroll
    for (int nf = 0; nf < 4; ++nf) acc[mi][nf] = zero8();

  issue_tile(0, 0);
  for (int t = 0; t < D_/32; ++t) {               // 32 k-steps
    const int s = t & 1;
    DRAIN_DS();
    __syncthreads();                              // prior reads of buf s^1 done
    if (t + 1 < D_/32) {
      issue_tile((t + 1)*32, s ^ 1);              // overlap DMA with compute
      WAIT_ASYNC(0x4);                            // tile t's 4 transfers landed
    } else {
      WAIT_ASYNC(0x0);
    }
    __syncthreads();                              // tile t visible to all waves

    v16bf af[2];
#pragma unroll
    for (int mi = 0; mi < 2; ++mi)
      af[mi] = load_a_frag(&ldsA[s][(32*wm + 16*mi)*LDA_], LDA_);
    v16bf bfr[4];
#pragma unroll
    for (int nf = 0; nf < 4; ++nf)
      bfr[nf] = load_b_frag(&ldsW[s][0], LDW_, 64*wn + nf*16);
#pragma unroll
    for (int nf = 0; nf < 4; ++nf)
#pragma unroll
      for (int mi = 0; mi < 2; ++mi)
        acc[mi][nf] = __builtin_amdgcn_wmma_f32_16x16x32_bf16(
            false, af[mi], false, bfr[nf], (short)0, acc[mi][nf], false, false);
  }

#pragma unroll
  for (int mi = 0; mi < 2; ++mi)
#pragma unroll
    for (int nf = 0; nf < 4; ++nf) {
      const int c = c0 + 64*wn + nf*16 + (lane & 15);
      const float bsv = bias[c];
#pragma unroll
      for (int i = 0; i < 8; ++i) {
        const int r = r0 + 32*wm + 16*mi + i + 8*(lane >> 4);
        const float val = acc[mi][nf][i] + bsv;
        if (mode == 0) {
          outF[(size_t)r*D_ + c] = val;
        } else {
          const int b = r >> 11, ss = r & (S_ - 1);
          const int h = c >> 6,  e = c & (HD_ - 1);
          if (mode == 1)
            outB[((size_t)(b*H_ + h)*S_ + ss)*HD_ + e] = to_bf16(val);
          else
            outB[((size_t)(b*H_ + h)*HD_ + e)*S_ + ss] = to_bf16(val);
        }
      }
    }
}

// Flash attention: one workgroup (4 waves) per (b,h,128-row block), online softmax.
// K/V tiles double-buffered through the async global->LDS DMA path (ASYNCcnt).
#define LDK_ 72     // 64 + 8 pad (144B rows)
__global__ __launch_bounds__(128) void attn_fwd(const __bf16* __restrict__ Qb,
                                                const __bf16* __restrict__ Kb,
                                                const __bf16* __restrict__ Vb,
                                                __bf16* __restrict__ Ob) {
  __shared__ __attribute__((aligned(16))) __bf16 ldsK[2][64*LDK_];   // [e][t]
  __shared__ __attribute__((aligned(16))) __bf16 ldsV[2][64*LDK_];   // [t][e]
  __shared__ __attribute__((aligned(16))) __bf16 ldsP[4][16*LDK_];
  const int tid  = threadIdx.x;
  const int lane = tid & 31;
  const int w    = tid >> 5;
  const int bh   = blockIdx.x >> 4;    // b*H + h
  const int rb   = blockIdx.x & 15;
  const int b    = bh >> 4, h = bh & 15;
  const int srow0 = rb*128 + 32*w;

  const __bf16* Qbase = Qb + (size_t)bh*S_*HD_;   // [s][e]
  const __bf16* Kbase = Kb + (size_t)bh*HD_*S_;   // [e][t]  (pre-transposed)
  const __bf16* Vbase = Vb + (size_t)bh*S_*HD_;   // [t][e]

  unsigned offK[2], offV[2];
#pragma unroll
  for (int s = 0; s < 2; ++s) {
    offK[s] = (unsigned)(size_t)(void*)&ldsK[s][0];
    offV[s] = (unsigned)(size_t)(void*)&ldsV[s][0];
  }

  // 8 async transfers per thread per key block (K tile + V tile, 64x64 each).
  auto issue_kv = [&](int kb, int s) {
#pragma unroll
    for (int it = 0; it < 4; ++it) {
      const int c = tid + it*128;                 // 0..511
      const int r = c >> 3, j = c & 7;
      const unsigned lk = offK[s] + (unsigned)((r*LDK_ + j*8)*2);
      const __bf16* gk = Kbase + (size_t)r*S_ + kb + j*8;
      ASYNC_B128(lk, gk);
      const unsigned lv = offV[s] + (unsigned)((r*LDK_ + j*8)*2);
      const __bf16* gv = Vbase + (size_t)(kb + r)*HD_ + j*8;
      ASYNC_B128(lv, gv);
    }
  };

  v16bf qf[2][2];
#pragma unroll
  for (int mi = 0; mi < 2; ++mi)
#pragma unroll
    for (int kf = 0; kf < 2; ++kf)
      qf[mi][kf] = load_a_frag(Qbase + (size_t)(srow0 + 16*mi)*HD_ + kf*32, HD_);

  v8f o[2][4];
  float mrun[2][8], lrun[2][8];
#pragma unroll
  for (int mi = 0; mi < 2; ++mi) {
#pragma unroll
    for (int nf = 0; nf < 4; ++nf) o[mi][nf] = zero8();
#pragma unroll
    for (int i = 0; i < 8; ++i) { mrun[mi][i] = -1e30f; lrun[mi][i] = 0.f; }
  }

  issue_kv(0, 0);
  for (int t = 0; t < S_/64; ++t) {               // 32 key blocks
    const int s = t & 1;
    DRAIN_DS();
    __syncthreads();                              // prior reads of buf s^1 done
    if (t + 1 < S_/64) {
      issue_kv((t + 1)*64, s ^ 1);                // overlap DMA with compute
      WAIT_ASYNC(0x8);                            // this block's 8 transfers landed
    } else {
      WAIT_ASYNC(0x0);
    }
    __syncthreads();

#pragma unroll
    for (int mi = 0; mi < 2; ++mi) {
      v8f sc[4];
#pragma unroll
      for (int nf = 0; nf < 4; ++nf) sc[nf] = zero8();
#pragma unroll
      for (int kf = 0; kf < 2; ++kf) {
#pragma unroll
        for (int nf = 0; nf < 4; ++nf) {
          v16bf bkf = load_b_frag(&ldsK[s][kf*32*LDK_], LDK_, nf*16);
          sc[nf] = __builtin_amdgcn_wmma_f32_16x16x32_bf16(
              false, qf[mi][kf], false, bkf, (short)0, sc[nf], false, false);
        }
      }
      // online softmax (scale = 1/sqrt(64) = 0.125); a row's 16 N-values live
      // in one 16-lane group -> butterfly over xor masks 1,2,4,8.
#pragma unroll
      for (int nf = 0; nf < 4; ++nf)
#pragma unroll
        for (int i = 0; i < 8; ++i) sc[nf][i] *= 0.125f;
#pragma unroll
      for (int i = 0; i < 8; ++i) {
        float vm = fmaxf(fmaxf(sc[0][i], sc[1][i]), fmaxf(sc[2][i], sc[3][i]));
        vm = fmaxf(vm, __shfl_xor(vm, 1, 32));
        vm = fmaxf(vm, __shfl_xor(vm, 2, 32));
        vm = fmaxf(vm, __shfl_xor(vm, 4, 32));
        vm = fmaxf(vm, __shfl_xor(vm, 8, 32));
        const float mnew = fmaxf(mrun[mi][i], vm);
        const float corr = __expf(mrun[mi][i] - mnew);
        mrun[mi][i] = mnew;
        float rs = 0.f;
#pragma unroll
        for (int nf = 0; nf < 4; ++nf) {
          const float p = __expf(sc[nf][i] - mnew);
          sc[nf][i] = p;
          rs += p;
        }
        rs += __shfl_xor(rs, 1, 32);
        rs += __shfl_xor(rs, 2, 32);
        rs += __shfl_xor(rs, 4, 32);
        rs += __shfl_xor(rs, 8, 32);
        lrun[mi][i] = lrun[mi][i]*corr + rs;
#pragma unroll
        for (int nf = 0; nf < 4; ++nf) o[mi][nf][i] *= corr;
      }
      // restage P (C layout -> A layout) through this wave's private LDS region
#pragma unroll
      for (int nf = 0; nf < 4; ++nf)
#pragma unroll
        for (int i = 0; i < 8; ++i)
          ldsP[w][(i + 8*(lane >> 4))*LDK_ + nf*16 + (lane & 15)] = to_bf16(sc[nf][i]);
      // O += P * V
#pragma unroll
      for (int kf = 0; kf < 2; ++kf) {
        v16bf pa = load_a_frag(&ldsP[w][0] + kf*32, LDK_);
#pragma unroll
        for (int nf = 0; nf < 4; ++nf) {
          v16bf bvf = load_b_frag(&ldsV[s][kf*32*LDK_], LDK_, nf*16);
          o[mi][nf] = __builtin_amdgcn_wmma_f32_16x16x32_bf16(
              false, pa, false, bvf, (short)0, o[mi][nf], false, false);
        }
      }
    }
  }

#pragma unroll
  for (int mi = 0; mi < 2; ++mi)
#pragma unroll
    for (int nf = 0; nf < 4; ++nf)
#pragma unroll
      for (int i = 0; i < 8; ++i) {
        const float val = o[mi][nf][i] / lrun[mi][i];
        const int srow = srow0 + 16*mi + i + 8*(lane >> 4);
        const int col  = h*HD_ + nf*16 + (lane & 15);
        Ob[((size_t)(b*S_ + srow))*D_ + col] = to_bf16(val);
      }
}

extern "C" void kernel_launch(void* const* d_in, const int* in_sizes, int n_in,
                              void* d_out, int out_size, void* d_ws, size_t ws_size,
                              hipStream_t stream) {
  (void)in_sizes; (void)n_in; (void)out_size; (void)ws_size;
  const float* x  = (const float*)d_in[0];
  const float* Wq = (const float*)d_in[1];
  const float* bq = (const float*)d_in[2];
  const float* Wk = (const float*)d_in[3];
  const float* bk = (const float*)d_in[4];
  const float* Wv = (const float*)d_in[5];
  const float* bv = (const float*)d_in[6];
  const float* Wp = (const float*)d_in[7];
  const float* bp = (const float*)d_in[8];

  char* ws = (char*)d_ws;
  const size_t MB = 1ull << 20;
  __bf16* Qb  = (__bf16*)(ws + 0*MB);    // [B,H,S,HD]  8 MiB
  __bf16* Kb  = (__bf16*)(ws + 8*MB);    // [B,H,HD,S]  8 MiB
  __bf16* Vb  = (__bf16*)(ws + 16*MB);   // [B,H,S,HD]  8 MiB
  __bf16* Ob  = (__bf16*)(ws + 24*MB);   // [B,S,D]     8 MiB
  __bf16* Wqt = (__bf16*)(ws + 32*MB);   // [in][out] bf16, 2 MiB each
  __bf16* Wkt = (__bf16*)(ws + 34*MB);
  __bf16* Wvt = (__bf16*)(ws + 36*MB);
  __bf16* Wpt = (__bf16*)(ws + 38*MB);
  __bf16* xb  = (__bf16*)(ws + 40*MB);   // [BS,D]      8 MiB

  cvt_f32_bf16<<<(BS_*D_ + 255)/256, 256, 0, stream>>>(x, xb, BS_*D_);
  dim3 tg(D_/32, D_/32);   // 32 x 32
  cvt_transpose_bf16<<<tg, 256, 0, stream>>>(Wq, Wqt);
  cvt_transpose_bf16<<<tg, 256, 0, stream>>>(Wk, Wkt);
  cvt_transpose_bf16<<<tg, 256, 0, stream>>>(Wv, Wvt);
  cvt_transpose_bf16<<<tg, 256, 0, stream>>>(Wp, Wpt);

  dim3 gg(BS_/128, D_/128);   // 32 x 8
  gemm_qkv<<<gg, 256, 0, stream>>>(xb, Wqt, bq, nullptr, Qb, 1);
  gemm_qkv<<<gg, 256, 0, stream>>>(xb, Wkt, bk, nullptr, Kb, 2);
  gemm_qkv<<<gg, 256, 0, stream>>>(xb, Wvt, bv, nullptr, Vb, 1);

  attn_fwd<<<B_*H_*(S_/128), 128, 0, stream>>>(Qb, Kb, Vb, Ob);

  gemm_qkv<<<gg, 256, 0, stream>>>(Ob, Wpt, bp, (float*)d_out, nullptr, 0);
}
